// InputEmbedder_68710886801424
// MI455X (gfx1250) — matrix operand order repaired and checked by reading, not compiled, not run on hardware
//
#include <hip/hip_runtime.h>
#include <hip/hip_bf16.h>
#include <math.h>

typedef __attribute__((ext_vector_type(16))) _Float16 v16h;
typedef __attribute__((ext_vector_type(8)))  _Float16 v8h;
typedef __attribute__((ext_vector_type(4)))  _Float16 v4h;
typedef __attribute__((ext_vector_type(8)))  float    v8f;

// ---- problem constants (from reference) ----
#define CHNLS   21
#define HIDDEN  128
#define BATCH   2
#define T_IN    15000
#define L1T     7497     // (15000-7)/2+1
#define L2T     3746
#define L3T     1870
#define L4T     932
#define TR      462      // (932-9)/2+1
#define C1      168
#define C2      336
#define C3      672
#define C4      1344
#define C5      2688

// workspace: two ping-pong regions (max layer = 2*168*7497 = 2,518,992 floats)
#define REGION_ELEMS 2621440u   // 10 MB each

// ---- fused-kernel LDS layout (bytes) ----
// A and B are stored pre-swizzled into WMMA fragment order:
//   As[mtile][lane][e]  (8  x 32 x 16 halves =  8192 B)
//   Bs[ntile][lane][e]  (32 x 32 x 16 halves = 32768 B)
#define A_OFF   0
#define B_OFF   8192
#define W_OFF   40960                   // f16 [65536] = 131072
#define S1_OFF  172032                  // f16 [32768] =  65536
#define S2_OFF  237568                  // f16 [16384] =  32768
#define S3_OFF  270336                  // f16 [8192]  =  16384
#define SMEM_BYTES 286720               // 280 KB < 320 KB/WGP

// ============================================================
// Phase 1: depthwise grouped conv stack (bandwidth-trivial)
// ============================================================

__global__ void dw_first(const float* __restrict__ x, const float* __restrict__ w,
                         const float* __restrict__ bias, float* __restrict__ out) {
  int idx = blockIdx.x * blockDim.x + threadIdx.x;
  const int total = BATCH * C1 * L1T;
  if (idx >= total) return;
  int t = idx % L1T;
  int c = (idx / L1T) % C1;
  int b = idx / (L1T * C1);
  int k = c >> 3;
  const float* wp = w + c * 7;
  const float* xp = x + ((size_t)b * T_IN + 2 * t) * CHNLS + k;
  float s = bias[c];
#pragma unroll
  for (int j = 0; j < 7; ++j) s += wp[j] * xp[j * CHNLS];
  out[((size_t)b * C1 + c) * L1T + t] = s;
}

__global__ void dw_generic(const float* __restrict__ in, const float* __restrict__ w,
                           const float* __restrict__ bias, float* __restrict__ out,
                           int Cin, int Cout, int Lin, int Lout, int K) {
  int idx = blockIdx.x * blockDim.x + threadIdx.x;
  int total = BATCH * Cout * Lout;
  if (idx >= total) return;
  int t = idx % Lout;
  int c = (idx / Lout) % Cout;
  int b = idx / (Lout * Cout);
  int g = c / (Cout / Cin);
  const float* ip = in + ((size_t)b * Cin + g) * Lin + 2 * t;
  const float* wp = w + c * K;
  float s = bias[c];
  for (int j = 0; j < K; ++j) s += wp[j] * ip[j];
  out[((size_t)b * Cout + c) * Lout + t] = s;
}

// ============================================================
// Phase 2: fused per-(b,t)  WMMA GEMM + LDS-resident conv/GELU chain
// ============================================================

__device__ __forceinline__ float gelu_exact(float v) {
  return 0.5f * v * (1.0f + erff(v * 0.70710678118654752f));
}

// k=4, stride=2, pad=1 conv over a flat f16 LDS vector; fp32 math.
// Each thread handles 4 consecutive outputs from one aligned v8h load + 2 halo loads.
__device__ __forceinline__ void conv_stage(const _Float16* __restrict__ in,
                                           _Float16* __restrict__ out,
                                           int Lin, int Lout,
                                           float w0, float w1, float w2, float w3,
                                           float bias, bool act, int tid) {
  const int nq = Lout >> 2;               // groups of 4 outputs
  for (int r = tid; r < nq; r += 256) {
    v8h mid = *(const v8h*)(in + 8 * r);  // in[8r .. 8r+7], 16B aligned
    float e[10];
    e[0] = (r == 0) ? 0.0f : (float)in[8 * r - 1];          // left halo / zero-pad
    e[9] = (8 * r + 8 < Lin) ? (float)in[8 * r + 8] : 0.0f; // right halo / zero-pad
#pragma unroll
    for (int i = 0; i < 8; ++i) e[i + 1] = (float)mid[i];
    v4h res;
#pragma unroll
    for (int q = 0; q < 4; ++q) {
      float s = bias + w0 * e[2 * q] + w1 * e[2 * q + 1]
                     + w2 * e[2 * q + 2] + w3 * e[2 * q + 3];
      if (act) s = gelu_exact(s);
      res[q] = (_Float16)s;
    }
    *(v4h*)(out + 4 * r) = res;           // 8B aligned
  }
}

__global__ void __launch_bounds__(256)
fused_gemm_chain(const float* __restrict__ emb,        // [B, 2688, TR]
                 const int* __restrict__ cv,           // [B, 21]
                 const float* __restrict__ ch_table,   // [21, 512]
                 const float* __restrict__ mw1, const float* __restrict__ mb1,
                 const float* __restrict__ mw2, const float* __restrict__ mb2,
                 const float* __restrict__ mw3, const float* __restrict__ mb3,
                 const float* __restrict__ mw4, const float* __restrict__ mb4,
                 float* __restrict__ out) {            // [B*TR, 4096] fp32
  extern __shared__ __align__(16) char smem[];
  _Float16* As = (_Float16*)(smem + A_OFF);
  _Float16* Bs = (_Float16*)(smem + B_OFF);
  _Float16* W  = (_Float16*)(smem + W_OFF);
  _Float16* S1 = (_Float16*)(smem + S1_OFF);
  _Float16* S2 = (_Float16*)(smem + S2_OFF);
  _Float16* S3 = (_Float16*)(smem + S3_OFF);

  const int tid = threadIdx.x;
  const int bt  = blockIdx.x;
  const int b   = bt / TR;
  const int t   = bt % TR;

  // hint: pull ch_table rows toward the caches (global_prefetch_b8)
  if (tid < CHNLS)
    __builtin_prefetch(ch_table + (size_t)cv[b * CHNLS + tid] * 512, 0, 0);

  // ---- stage A = emb[b, i*21+k, t] into WMMA-fragment-swizzled LDS ----
  // fragment mapping: K -> (h = (K>>3)&1, e = (K&7) + ((K>>4)<<3)), lane = h*16 + (i&15)
  for (int idx = tid; idx < 128 * 32; idx += 256) {
    int i = idx >> 5, K = idx & 31;
    float v = (K < CHNLS) ? emb[((size_t)b * C5 + i * CHNLS + K) * TR + t] : 0.0f;
    int h = (K >> 3) & 1;
    int e = (K & 7) + ((K >> 4) << 3);
    As[(i >> 4) * 512 + (h * 16 + (i & 15)) * 16 + e] = (_Float16)v;
  }
  // ---- stage B = ch_table[ch_vector[b,k]] into fragment-swizzled LDS ----
  // fragment mapping: k -> (h = (k>>3)&1, e = (k&7) + ((k>>4)<<3)), lane = h*16 + (j&15)
  for (int idx = tid; idx < (32 * 512) / 4; idx += 256) {
    int k  = idx >> 7;            // 0..31
    int j4 = (idx & 127) << 2;    // 0..508, step 4
    float4 v = make_float4(0.f, 0.f, 0.f, 0.f);
    if (k < CHNLS)
      v = *(const float4*)(ch_table + (size_t)cv[b * CHNLS + k] * 512 + j4);
    int h = (k >> 3) & 1, e = (k & 7) + ((k >> 4) << 3);
#pragma unroll
    for (int q = 0; q < 4; ++q) {
      int j = j4 + q;
      Bs[(j >> 4) * 512 + (h * 16 + (j & 15)) * 16 + e] = (_Float16)((&v.x)[q]);
    }
  }
  __syncthreads();

  // ---- WMMA GEMM: W[128,512] = A[128,32] x B[32,512], fp32 accumulate ----
  {
    const int lane  = tid & 31;
    const int wave  = tid >> 5;     // 8 waves -> 8 M-tiles
    const int h     = lane >> 4;
    const int ln    = lane & 15;
    const int mtile = wave;

    // whole 32-byte fragment per lane: two ds_load_b128
    v16h a = *(const v16h*)(As + mtile * 512 + lane * 16);

    for (int ntile = 0; ntile < 32; ++ntile) {
      v16h bf = *(const v16h*)(Bs + ntile * 512 + lane * 16);
      v8f c = {};
      c = __builtin_amdgcn_wmma_f32_16x16x32_f16(
              /*neg_a=*/false, a, /*neg_b=*/false, bf,
              /*c_mod=*/(short)0, c, /*reuse_a=*/false, /*reuse_b=*/false);
      // D layout: VGPR v -> M = v + 8*h, N = ln
#pragma unroll
      for (int v = 0; v < 8; ++v)
        W[(mtile * 16 + v + 8 * h) * 512 + ntile * 16 + ln] = (_Float16)c[v];
    }
  }
  __syncthreads();

  // ---- per-(b,t) conv chain over flat 65536 vector, entirely in LDS ----
  conv_stage(W,  S1, 65536, 32768, mw1[0], mw1[1], mw1[2], mw1[3], mb1[0], false, tid);
  __syncthreads();
  conv_stage(S1, S2, 32768, 16384, mw2[0], mw2[1], mw2[2], mw2[3], mb2[0], true,  tid);
  __syncthreads();
  conv_stage(S2, S3, 16384, 8192,  mw3[0], mw3[1], mw3[2], mw3[3], mb3[0], false, tid);
  __syncthreads();

  // final stage: gelu, vectorized fp32 stores to global
  {
    float w0 = mw4[0], w1 = mw4[1], w2 = mw4[2], w3 = mw4[3], bb = mb4[0];
    float* op = out + (size_t)bt * 4096;
    for (int r = tid; r < 4096 / 4; r += 256) {
      v8h mid = *(const v8h*)(S3 + 8 * r);
      float e[10];
      e[0] = (r == 0) ? 0.0f : (float)S3[8 * r - 1];
      e[9] = (8 * r + 8 < 8192) ? (float)S3[8 * r + 8] : 0.0f;
#pragma unroll
      for (int i = 0; i < 8; ++i) e[i + 1] = (float)mid[i];
      float4 o4;
#pragma unroll
      for (int q = 0; q < 4; ++q) {
        float s = bb + w0 * e[2 * q] + w1 * e[2 * q + 1]
                     + w2 * e[2 * q + 2] + w3 * e[2 * q + 3];
        (&o4.x)[q] = gelu_exact(s);
      }
      *(float4*)(op + 4 * r) = o4;  // 16B aligned
    }
  }
}

// ============================================================
// host launch
// ============================================================
static inline int cdiv_i(int a, int b) { return (a + b - 1) / b; }

extern "C" void kernel_launch(void* const* d_in, const int* in_sizes, int n_in,
                              void* d_out, int out_size, void* d_ws, size_t ws_size,
                              hipStream_t stream) {
  (void)in_sizes; (void)n_in; (void)out_size; (void)ws_size;

  const float* x        = (const float*)d_in[0];
  const int*   cv       = (const int*)  d_in[1];
  const float* w1 = (const float*)d_in[2];  const float* b1 = (const float*)d_in[3];
  const float* w2 = (const float*)d_in[4];  const float* b2 = (const float*)d_in[5];
  const float* w3 = (const float*)d_in[6];  const float* b3 = (const float*)d_in[7];
  const float* w4 = (const float*)d_in[8];  const float* b4 = (const float*)d_in[9];
  const float* w5 = (const float*)d_in[10]; const float* b5 = (const float*)d_in[11];
  const float* ch_table = (const float*)d_in[12];
  const float* mw1 = (const float*)d_in[13]; const float* mb1 = (const float*)d_in[14];
  const float* mw2 = (const float*)d_in[15]; const float* mb2 = (const float*)d_in[16];
  const float* mw3 = (const float*)d_in[17]; const float* mb3 = (const float*)d_in[18];
  const float* mw4 = (const float*)d_in[19]; const float* mb4 = (const float*)d_in[20];

  float* r0 = (float*)d_ws;
  float* r1 = r0 + REGION_ELEMS;

  // depthwise stack: x -> r0 -> r1 -> r0 -> r1 -> r0 (= emb [B,2688,TR])
  dw_first  <<<cdiv_i(BATCH * C1 * L1T, 256), 256, 0, stream>>>(x, w1, b1, r0);
  dw_generic<<<cdiv_i(BATCH * C2 * L2T, 256), 256, 0, stream>>>(r0, w2, b2, r1, C1, C2, L1T, L2T, 7);
  dw_generic<<<cdiv_i(BATCH * C3 * L3T, 256), 256, 0, stream>>>(r1, w3, b3, r0, C2, C3, L2T, L3T, 7);
  dw_generic<<<cdiv_i(BATCH * C4 * L4T, 256), 256, 0, stream>>>(r0, w4, b4, r1, C3, C4, L3T, L4T, 7);
  dw_generic<<<cdiv_i(BATCH * C5 * TR,  256), 256, 0, stream>>>(r1, w5, b5, r0, C4, C5, L4T, TR, 9);

  // fused WMMA GEMM + conv chain: one workgroup per (b,t), 280 KB dynamic LDS
  hipFuncSetAttribute(reinterpret_cast<const void*>(fused_gemm_chain),
                      hipFuncAttributeMaxDynamicSharedMemorySize, SMEM_BYTES);
  fused_gemm_chain<<<BATCH * TR, 256, SMEM_BYTES, stream>>>(
      r0, cv, ch_table, mw1, mb1, mw2, mb2, mw3, mb3, mw4, mb4, (float*)d_out);
}
